// BetterLSTM_56504589746224
// MI455X (gfx1250) — compile-verified, matrix operand order checked by reading, not run on hardware
//
#include <hip/hip_runtime.h>
#include <hip/hip_bf16.h>
#include <stdint.h>

// ---------------------------------------------------------------------------
// Fused persistent LSTM for MI455X (gfx1250, wave32, WMMA, async-to-LDS).
//   - bf16 WMMA 16x16x32, fp32 accumulate
//   - 32 persistent workgroups, each owns 16 hidden units (64 gate columns)
//   - W_ih / W_hh slices pre-packed into LDS in WMMA B-fragment order
//   - h_{t-1} staged into LDS via GLOBAL_LOAD_ASYNC_TO_LDS_B128, overlapped
//     with the x-projection GEMM; single s_wait_asynccnt before the h-GEMM
//   - c state lives in LDS for the whole scan; h ping-pongs through global
//   - one device-wide barrier per timestep (atomic arrive + s_sleep spin)
// ---------------------------------------------------------------------------

#define B_   64
#define T_   1024
#define F_   256
#define H_   512
#define G4H  2048      // 4*H
#define NWG  32        // persistent workgroups (co-resident, 1 per WGP)
#define NTHREADS 256   // 8 waves

typedef __attribute__((ext_vector_type(16))) __bf16 v16bf;
typedef __attribute__((ext_vector_type(8)))  float  v8f;

union Frag32B { uint4 q[2]; v16bf v; };   // 32 bytes = 16 bf16 per lane

__device__ __forceinline__ unsigned short f2bf_bits(float f) {
  union { float f; unsigned u; } v; v.f = f;
  unsigned r = v.u + 0x7FFFu + ((v.u >> 16) & 1u);   // round-to-nearest-even
  return (unsigned short)(r >> 16);
}
__device__ __forceinline__ float sigf(float x) { return 1.f / (1.f + __expf(-x)); }
__device__ __forceinline__ float tanh_fast(float x) {
  return 1.f - 2.f / (__expf(2.f * x) + 1.f);   // saturates correctly at +-inf
}

// ---------------------------------------------------------------------------
__global__ void init_counters_kernel(unsigned* c) {
  if (threadIdx.x < 8) c[threadIdx.x] = 0u;
}

// Teacher-forcing shift + fp32 -> bf16 conversion of the input sequence.
__global__ void shift_convert_kernel(const float* __restrict__ x,
                                     const int* __restrict__ train_mode,
                                     unsigned short* __restrict__ xseq,
                                     long n) {
  const int tm = train_mode[0];
  for (long i = (long)blockIdx.x * blockDim.x + threadIdx.x; i < n;
       i += (long)gridDim.x * blockDim.x) {
    long bt = i / F_;
    long t  = bt % T_;
    float v = tm ? ((t == 0) ? 0.f : x[i - F_]) : x[i];
    xseq[i] = f2bf_bits(v);
  }
}

// ---------------------------------------------------------------------------
// One LSTM layer, full T-scan, persistent grid of NWG blocks.
// ---------------------------------------------------------------------------
__global__ __launch_bounds__(NTHREADS)
void lstm_layer_persistent(const unsigned short* __restrict__ in_seq, int Kin,
                           const float* __restrict__ W_ih,   // [4H, Kin]
                           const float* __restrict__ W_hh,   // [4H, H]
                           const float* __restrict__ b_ih,
                           const float* __restrict__ b_hh,
                           unsigned short* __restrict__ hbuf,   // [2,B,H] bf16
                           unsigned short* __restrict__ out_seq,// [B,T,H] bf16 or 0
                           float* __restrict__ out_f32,         // [B,T,H] f32 or 0
                           unsigned* __restrict__ counter) {
  extern __shared__ char smem[];
  // LDS carve: h stage | whh pack | wih pack | gates | c | bias
  unsigned short* ldsH  = (unsigned short*)smem;                 // B*H bf16 = 64KB
  unsigned short* whhP  = ldsH + (size_t)B_ * H_;                // 64KB
  unsigned short* wihP  = whhP + 4 * (H_ / 32) * 32 * 16;        // Kin*128 B
  float*          gatesL = (float*)(wihP + 4 * (Kin / 32) * 32 * 16);
  float*          cL     = gatesL + 64 * 64;
  float*          biasL  = cL + 64 * 16;

  const int tid  = threadIdx.x;
  const int lane = tid & 31;
  const int wave = tid >> 5;
  const int hb   = blockIdx.x * (H_ / NWG);     // first hidden unit owned

  // ---- phase 0: pack weight slices into LDS in WMMA B-fragment order ----
  //   fidx = n_tile*(Kw/32)+kk ; lane: N=lane%16, K = kk*32 + {8h..8h+7,16+8h..}
  {
    const int half = lane >> 4;
    const int nl   = lane & 15;
    for (int f = wave; f < 4 * (H_ / 32); f += 8) {
      int n_tile = f / (H_ / 32), kk = f % (H_ / 32);
      int r  = n_tile * H_ + hb + nl;           // global gate row (gate==n_tile)
      int c0 = kk * 32 + 8 * half;
      const float* src = W_hh + (size_t)r * H_;
      unsigned short* dst = whhP + ((size_t)f * 32 + lane) * 16;
#pragma unroll
      for (int i = 0; i < 8; ++i) dst[i]     = f2bf_bits(src[c0 + i]);
#pragma unroll
      for (int i = 0; i < 8; ++i) dst[8 + i] = f2bf_bits(src[c0 + 16 + i]);
    }
    const int nkk = Kin / 32;
    for (int f = wave; f < 4 * nkk; f += 8) {
      int n_tile = f / nkk, kk = f % nkk;
      int r  = n_tile * H_ + hb + nl;
      int c0 = kk * 32 + 8 * half;
      const float* src = W_ih + (size_t)r * Kin;
      unsigned short* dst = wihP + ((size_t)f * 32 + lane) * 16;
#pragma unroll
      for (int i = 0; i < 8; ++i) dst[i]     = f2bf_bits(src[c0 + i]);
#pragma unroll
      for (int i = 0; i < 8; ++i) dst[8 + i] = f2bf_bits(src[c0 + 16 + i]);
    }
  }
  if (tid < 64) {
    int r = (tid >> 4) * H_ + hb + (tid & 15);
    biasL[tid] = b_ih[r] + b_hh[r];
  }
  for (int e = tid; e < 64 * 16; e += NTHREADS) cL[e] = 0.f;
  __syncthreads();

  // wave -> C-tile assignment: 64x64 tile = 4x4 of 16x16; 2 N-tiles per wave
  const int m_t    = wave & 3;
  const int ntBase = (wave >> 2) * 2;
  const int half   = lane >> 4;
  const int mRow   = m_t * 16 + (lane & 15);    // A row this lane feeds
  const int nkkI   = Kin / 32;
  const int nkkH   = H_ / 32;
  const unsigned ldsHBase = (unsigned)(size_t)ldsH;

  for (int t = 0; t < T_; ++t) {
    // ---- kick off async staging of h_{t-1} into LDS (overlaps GEMM1) ----
    // per lane: 16B global -> 16B LDS, ASYNCcnt-tracked
    if (t > 0) {
      const char* hPrev = (const char*)(hbuf + (size_t)((t + 1) & 1) * B_ * H_);
#pragma unroll
      for (int it = 0; it < (B_ * H_ * 2) / (NTHREADS * 16); ++it) {  // 16 iters
        unsigned byteOff = (unsigned)(it * NTHREADS + tid) * 16u;
        unsigned ldsAddr = ldsHBase + byteOff;
        const void* g = hPrev + byteOff;
        asm volatile("global_load_async_to_lds_b128 %0, %1, off"
                     :: "v"(ldsAddr), "v"(g) : "memory");
      }
    }

    v8f acc0 = {}, acc1 = {};

    // ---- GEMM1: x_t @ W_ih_slice  (A from global, B from LDS pack) ----
    {
      const unsigned short* aRow = in_seq + ((size_t)mRow * T_ + t) * Kin;
      for (int kk = 0; kk < nkkI; ++kk) {
        Frag32B a, b0, b1;
        int c0 = kk * 32 + 8 * half;
        a.q[0] = *(const uint4*)(aRow + c0);
        a.q[1] = *(const uint4*)(aRow + c0 + 16);
        const uint4* p0 = (const uint4*)(wihP + ((size_t)((ntBase)     * nkkI + kk) * 32 + lane) * 16);
        const uint4* p1 = (const uint4*)(wihP + ((size_t)((ntBase + 1) * nkkI + kk) * 32 + lane) * 16);
        b0.q[0] = p0[0]; b0.q[1] = p0[1];
        b1.q[0] = p1[0]; b1.q[1] = p1[1];
        acc0 = __builtin_amdgcn_wmma_f32_16x16x32_bf16(false, a.v, false, b0.v, (short)0, acc0, false, false);
        acc1 = __builtin_amdgcn_wmma_f32_16x16x32_bf16(false, a.v, false, b1.v, (short)0, acc1, false, false);
      }
    }

    // prefetch next timestep's x rows into cache (global_prefetch_b8)
    if (t + 1 < T_) __builtin_prefetch(in_seq + ((size_t)mRow * T_ + (t + 1)) * Kin, 0, 1);

    // ---- GEMM2: h_{t-1} @ W_hh_slice, A fragments from LDS stage ----
    if (t > 0) {
      asm volatile("s_wait_asynccnt 0" ::: "memory");
      __syncthreads();   // LDS h stage visible to all waves
      const unsigned short* aRow = ldsH + (size_t)mRow * H_;
      for (int kk = 0; kk < nkkH; ++kk) {
        Frag32B a, b0, b1;
        int c0 = kk * 32 + 8 * half;
        a.q[0] = *(const uint4*)(aRow + c0);
        a.q[1] = *(const uint4*)(aRow + c0 + 16);
        const uint4* p0 = (const uint4*)(whhP + ((size_t)((ntBase)     * nkkH + kk) * 32 + lane) * 16);
        const uint4* p1 = (const uint4*)(whhP + ((size_t)((ntBase + 1) * nkkH + kk) * 32 + lane) * 16);
        b0.q[0] = p0[0]; b0.q[1] = p0[1];
        b1.q[0] = p1[0]; b1.q[1] = p1[1];
        acc0 = __builtin_amdgcn_wmma_f32_16x16x32_bf16(false, a.v, false, b0.v, (short)0, acc0, false, false);
        acc1 = __builtin_amdgcn_wmma_f32_16x16x32_bf16(false, a.v, false, b1.v, (short)0, acc1, false, false);
      }
    }

    // ---- C tiles (+bias) -> gates LDS ----
    // D layout: vgpr r, lanes 0-15: M=r, N=lane; lanes 16-31: M=8+r, N=lane-16
    {
      int n0 = ntBase * 16 + (lane & 15);
      int n1 = n0 + 16;
      int mo = m_t * 16 + 8 * half;
      float bb0 = biasL[n0], bb1 = biasL[n1];
#pragma unroll
      for (int r = 0; r < 8; ++r) {
        gatesL[(mo + r) * 64 + n0] = acc0[r] + bb0;
        gatesL[(mo + r) * 64 + n1] = acc1[r] + bb1;
      }
    }
    __syncthreads();

    // ---- gate nonlinearity + state update + h publish ----
    for (int e = tid; e < 64 * 16; e += NTHREADS) {
      int m = e >> 4, hid = e & 15;
      float gi = gatesL[m * 64 + hid];
      float gf = gatesL[m * 64 + 16 + hid];
      float gg = gatesL[m * 64 + 32 + hid];
      float go = gatesL[m * 64 + 48 + hid];
      float c  = sigf(gf) * cL[e] + sigf(gi) * tanh_fast(gg);
      float h  = sigf(go) * tanh_fast(c);
      cL[e] = c;
      int gcol = hb + hid;
      hbuf[((size_t)(t & 1) * B_ + m) * H_ + gcol] = f2bf_bits(h);
      if (out_seq) out_seq[((size_t)m * T_ + t) * H_ + gcol] = f2bf_bits(h);
      if (out_f32) out_f32[((size_t)m * T_ + t) * H_ + gcol] = h;
    }

    // ---- device-wide barrier: h_t published before any WG starts t+1 ----
    __syncthreads();
    if (tid == 0) {
      __threadfence();
      atomicAdd(counter, 1u);
      const unsigned target = (unsigned)NWG * (unsigned)(t + 1);
      while (__hip_atomic_load(counter, __ATOMIC_RELAXED, __HIP_MEMORY_SCOPE_AGENT) < target) {
        __builtin_amdgcn_s_sleep(1);
      }
    }
    __syncthreads();
  }
}

// ---------------------------------------------------------------------------
extern "C" void kernel_launch(void* const* d_in, const int* in_sizes, int n_in,
                              void* d_out, int out_size, void* d_ws, size_t ws_size,
                              hipStream_t stream) {
  const float* x     = (const float*)d_in[0];
  const float* W_ih0 = (const float*)d_in[1];
  const float* W_hh0 = (const float*)d_in[2];
  const float* b_ih0 = (const float*)d_in[3];
  const float* b_hh0 = (const float*)d_in[4];
  const float* W_ihR = (const float*)d_in[5];  // [3, 4H, H]
  const float* W_hhR = (const float*)d_in[6];  // [3, 4H, H]
  const float* b_ihR = (const float*)d_in[7];  // [3, 4H]
  const float* b_hhR = (const float*)d_in[8];  // [3, 4H]
  const int*   train = (const int*)d_in[9];
  float* out = (float*)d_out;
  (void)in_sizes; (void)n_in; (void)out_size; (void)ws_size;

  // workspace carve (~160 MB): counters | xseq0 bf16 | hseqA | hseqB | h ping-pong
  char* ws = (char*)d_ws;
  unsigned*       counters = (unsigned*)ws;
  unsigned short* xseq0 = (unsigned short*)(ws + 256);
  unsigned short* hseqA = xseq0 + (size_t)B_ * T_ * F_;
  unsigned short* hseqB = hseqA + (size_t)B_ * T_ * H_;
  unsigned short* hping = hseqB + (size_t)B_ * T_ * H_;

  auto smemFor = [](int Kin) -> size_t {
    return (size_t)B_ * H_ * 2                   // h stage
         + (size_t)(4 * (H_ / 32) * 32 * 32)     // whh pack
         + (size_t)(4 * (Kin / 32) * 32 * 32)    // wih pack
         + 64 * 64 * 4 + 64 * 16 * 4 + 64 * 4;   // gates + c + bias
  };
  // unconditional (deterministic, capture-safe; no static guards allowed)
  hipFuncSetAttribute((const void*)lstm_layer_persistent,
                      hipFuncAttributeMaxDynamicSharedMemorySize, (int)smemFor(H_));

  init_counters_kernel<<<1, 32, 0, stream>>>(counters);
  shift_convert_kernel<<<2048, 256, 0, stream>>>(x, train, xseq0, (long)B_ * T_ * F_);

  // layer 0: K=F, reads xseq0, writes hseqA
  lstm_layer_persistent<<<NWG, NTHREADS, smemFor(F_), stream>>>(
      xseq0, F_, W_ih0, W_hh0, b_ih0, b_hh0, hping, hseqA, nullptr, counters + 0);
  // layer 1: hseqA -> hseqB
  lstm_layer_persistent<<<NWG, NTHREADS, smemFor(H_), stream>>>(
      hseqA, H_, W_ihR + 0 * (size_t)G4H * H_, W_hhR + 0 * (size_t)G4H * H_,
      b_ihR + 0 * G4H, b_hhR + 0 * G4H, hping, hseqB, nullptr, counters + 1);
  // layer 2: hseqB -> hseqA
  lstm_layer_persistent<<<NWG, NTHREADS, smemFor(H_), stream>>>(
      hseqB, H_, W_ihR + 1 * (size_t)G4H * H_, W_hhR + 1 * (size_t)G4H * H_,
      b_ihR + 1 * G4H, b_hhR + 1 * G4H, hping, hseqA, nullptr, counters + 2);
  // layer 3: hseqA -> d_out (fp32)
  lstm_layer_persistent<<<NWG, NTHREADS, smemFor(H_), stream>>>(
      hseqA, H_, W_ihR + 2 * (size_t)G4H * H_, W_hhR + 2 * (size_t)G4H * H_,
      b_ihR + 2 * G4H, b_hhR + 2 * G4H, hping, nullptr, out, counters + 3);
}
